// TextEmbedding_26843545600239
// MI455X (gfx1250) — matrix operand order repaired and checked by reading, not compiled, not run on hardware
//
#include <hip/hip_runtime.h>
#include <math.h>

// ---------------- fixed problem geometry (from reference) ----------------
constexpr int kB   = 8;
constexpr int kNT  = 2048;
constexpr int kSEQ = 4096;
constexpr int kD   = 512;
constexpr int kDI  = 1024;
constexpr int kL   = 4;
constexpr int kM   = kB * kSEQ;          // 32768 rows for the GEMMs
constexpr int kMAXPOS = 4096;

// ---------------- CDNA5 WMMA vector types ----------------
typedef __attribute__((ext_vector_type(16))) __bf16       v16bf;
typedef __attribute__((ext_vector_type(8)))  float        v8f;
typedef __attribute__((ext_vector_type(8)))  unsigned int v8u;

__device__ __forceinline__ unsigned short f2bf(float f) {
  unsigned int u = __float_as_uint(f);
  u += 0x7FFFu + ((u >> 16) & 1u);          // round-to-nearest-even
  return (unsigned short)(u >> 16);
}
__device__ __forceinline__ float bf2f(unsigned short h) {
  return __uint_as_float(((unsigned int)h) << 16);
}
__device__ __forceinline__ float gelu_exact(float v) {
  return 0.5f * v * (1.0f + erff(v * 0.70710678118654752f));
}

// ---------------- CDNA5 async Global->LDS helpers ----------------
__device__ __forceinline__ unsigned int lds_off(const void* p) {
  // generic 'shared' pointer: low 32 bits are the LDS byte offset
  return (unsigned int)(unsigned long long)p;
}
__device__ __forceinline__ void async_copy_b128(void* lds_dst,
                                                const unsigned short* gsrc) {
  // GLOBAL_LOAD_ASYNC_TO_LDS_B128: per-lane 16B Global -> LDS, tracked by ASYNCcnt
  asm volatile("global_load_async_to_lds_b128 %0, %1, off"
               :: "v"(lds_off(lds_dst)), "v"(gsrc)
               : "memory");
}
__device__ __forceinline__ void wait_async0() {
#if defined(__has_builtin) && __has_builtin(__builtin_amdgcn_s_wait_asynccnt)
  __builtin_amdgcn_s_wait_asynccnt(0);
#else
  asm volatile("s_wait_asynccnt 0x0" ::: "memory");
#endif
}

// A fragment (16x32 bf16, row-major source, ld = K):
// lane L holds row M = L%16; elems 0..7 -> K = kb+g0+e, elems 8..15 -> K = kb+16+g0+(e-8)
__device__ __forceinline__ v16bf load_frag_a(const unsigned short* __restrict__ arow,
                                             int lane, int kb) {
  const int g0 = (lane >= 16) ? 8 : 0;
  const unsigned short* p = arow + kb + g0;
  uint4 lo = *reinterpret_cast<const uint4*>(p);
  uint4 hi = *reinterpret_cast<const uint4*>(p + 16);
  v8u r = { lo.x, lo.y, lo.z, lo.w, hi.x, hi.y, hi.z, hi.w };
  return __builtin_bit_cast(v16bf, r);
}

// B fragment (32x16 bf16) out of the LDS panel (row stride = BSTRIDE ushorts):
// lane L holds column N = L%16; elems 0..7 -> K = c0+e, elems 8..15 -> K = 16+c0+(e-8) (c0 = lane>=16 ? 8 : 0)
constexpr int kBSTRIDE = 40;  // 32 K-ushorts + 8 pad (80B rows: 16B-aligned, bank-spread)

__device__ __forceinline__ v16bf load_frag_b_lds(const unsigned short* panel,
                                                 int colLocal, int lane) {
  const int c0 = (lane >= 16) ? 8 : 0;
  const unsigned short* p = panel + colLocal * kBSTRIDE + c0;
  uint4 lo = *reinterpret_cast<const uint4*>(p);
  uint4 hi = *reinterpret_cast<const uint4*>(p + 16);
  v8u r = { lo.x, lo.y, lo.z, lo.w, hi.x, hi.y, hi.z, hi.w };
  return __builtin_bit_cast(v16bf, r);
}

// ---------------- WMMA GEMM ----------------
// Workgroup = 256 threads (8 waves): 128 (M) x 128 (N) tile.
// B panel (128 cols x 32 K bf16) async-staged Global->LDS, double-buffered,
// shared by all 8 waves. Each wave owns 16 rows x all 128 cols (8 accumulators).
// A fragments software-pipelined one K-step ahead; B fragments pipelined
// 2-deep through a rolling 3-register window so DS latency hides under WMMAs.
// GELU_OUT=true : out_bf = bf16(gelu(A*B + bias))                     (GEMM1)
// GELU_OUT=false: out_f32 = (A*B + bias + residual), pad rows zeroed  (GEMM2, in-place x)
template <int K, int N, bool GELU_OUT>
__global__ __launch_bounds__(256)
void wmma_gemm_kernel(const unsigned short* __restrict__ A,
                      const unsigned short* __restrict__ Bt,
                      const float* __restrict__ bias,
                      unsigned short* __restrict__ out_bf,
                      float* __restrict__ out_f32,
                      const float* __restrict__ residual,
                      const int* __restrict__ seq_len_ptr) {
  __shared__ unsigned short bstage[2][128 * kBSTRIDE];   // 2 x 10KB

  const int tid  = threadIdx.x;
  const int lane = tid & 31;
  const int wave = tid >> 5;                     // 0..7
  const int mt   = blockIdx.x * 8 + wave;        // this wave's 16-row M tile
  const int nt0  = blockIdx.y * 8;               // 128-col N strip (8 x 16)

  // staging map: 512 chunks of 16B (128 cols x 4 chunks); thread t copies
  // chunk t (col = t>>2) and chunk t+256 (col = t>>2 + 64), koff = (t&3)*8
  const int scol  = tid >> 2;
  const int skoff = (tid & 3) * 8;
  const unsigned short* bsrc0 =
      Bt + (size_t)(blockIdx.y * 128 + scol) * K + skoff;
  const unsigned short* bsrc1 =
      Bt + (size_t)(blockIdx.y * 128 + scol + 64) * K + skoff;
  unsigned short* sdst0[2] = { &bstage[0][scol * kBSTRIDE + skoff],
                               &bstage[1][scol * kBSTRIDE + skoff] };
  unsigned short* sdst1[2] = { &bstage[0][(scol + 64) * kBSTRIDE + skoff],
                               &bstage[1][(scol + 64) * kBSTRIDE + skoff] };

  const unsigned short* arow = A + (size_t)(mt * 16 + (lane & 15)) * K;

  // prologue: stage K-step 0 into buffer 0
  async_copy_b128(sdst0[0], bsrc0);
  async_copy_b128(sdst1[0], bsrc1);
  wait_async0();
  __syncthreads();

  v8f acc[8] = {};
  v16bf a = load_frag_a(arow, lane, 0);
  const int colb = lane & 15;
  for (int kb = 0; kb < K; kb += 32) {
    const int cur = (kb >> 5) & 1;
    const int nxt = cur ^ 1;
    if (kb + 32 < K) {                   // stage next B panel into other buffer
      async_copy_b128(sdst0[nxt], bsrc0 + kb + 32);
      async_copy_b128(sdst1[nxt], bsrc1 + kb + 32);
    }
    if (kb + 64 < K)                     // prefetch A two K-steps ahead
      __builtin_prefetch(arow + kb + 64, 0, 1);

    // pipeline: next A fragment in flight while WMMAs run on current one
    v16bf a_next = a;
    if (kb + 32 < K) a_next = load_frag_a(arow, lane, kb + 32);

    const unsigned short* panel = bstage[cur];
    // rolling 3-register B pipeline: loads for frags j+1, j+2 in flight at WMMA j
    v16bf bA = load_frag_b_lds(panel, 0 * 16 + colb, lane);
    v16bf bB = load_frag_b_lds(panel, 1 * 16 + colb, lane);
#pragma unroll
    for (int j = 0; j < 8; ++j) {
      v16bf bC = (j + 2 < 8) ? load_frag_b_lds(panel, (j + 2) * 16 + colb, lane)
                             : bA;
      acc[j] = __builtin_amdgcn_wmma_f32_16x16x32_bf16(
          /*neg_a=*/false, a, /*neg_b=*/false, bA,
          /*c_mod=*/(short)0, acc[j], /*reuse_a=*/false, /*reuse_b=*/false);
      bA = bB; bB = bC;
    }

    if (kb + 32 < K) wait_async0();      // next panel fully in LDS
    __syncthreads();                     // all waves done reading 'cur'
    a = a_next;
  }

  // C/D layout: VGPR r -> row = r + 8*(lane>=16), col = lane%16
  const int col   = lane & 15;
  const int rbase = mt * 16 + ((lane >= 16) ? 8 : 0);
  const int valid_len = min(kNT, seq_len_ptr[0]);
#pragma unroll
  for (int j = 0; j < 8; ++j) {
    const int cc = (nt0 + j) * 16 + col;
    const float bv = bias[cc];
#pragma unroll
    for (int r = 0; r < 8; ++r) {
      const int rr = rbase + r;
      float v = acc[j][r] + bv;
      if constexpr (GELU_OUT) {
        out_bf[(size_t)rr * N + cc] = f2bf(gelu_exact(v));
      } else {
        v += residual[(size_t)rr * N + cc];
        const int p = rr & (kSEQ - 1);
        out_f32[(size_t)rr * N + cc] = (p < valid_len) ? v : 0.0f;
      }
    }
  }
}

// ---------------- embedding + positional encoding ----------------
__global__ __launch_bounds__(256)
void embed_kernel(const int* __restrict__ text, const float* __restrict__ table,
                  const int* __restrict__ seq_len_ptr, float* __restrict__ x) {
  const size_t idx = (size_t)blockIdx.x * 256 + threadIdx.x;   // over kM*kD
  const int c = (int)(idx % kD);
  const size_t m = idx / kD;
  const int p = (int)(m % kSEQ);
  const int b = (int)(m / kSEQ);
  const int valid_len = min(kNT, seq_len_ptr[0]);
  float v = 0.0f;
  if (p < valid_len) {
    const int tok = text[(size_t)b * kNT + p] + 1;
    v = table[(size_t)tok * kD + c];
    const int i = c & (kD / 2 - 1);
    const float freq = __powf(10000.0f, -(float)i * (2.0f / (float)kD));
    const float f = (float)min(p, kMAXPOS - 1) * freq;
    v += (c < kD / 2) ? __cosf(f) : __sinf(f);
  }
  x[idx] = v;
}

// ---------------- depthwise conv(7) + LayerNorm -> bf16 GEMM operand ----------------
__global__ __launch_bounds__(256)
void dwconv_ln_kernel(const float* __restrict__ x, const float* __restrict__ w_dw,
                      const float* __restrict__ b_dw, const float* __restrict__ g,
                      const float* __restrict__ beta, unsigned short* __restrict__ yA) {
  const int m = blockIdx.x;               // row in [0, kM)
  const int p = m % kSEQ;
  const int b = m / kSEQ;
  const int t = threadIdx.x;

  float vals[2];
  float s = 0.f, s2 = 0.f;
#pragma unroll
  for (int h = 0; h < 2; ++h) {
    const int c = t + h * 256;
    float acc = b_dw[c];
#pragma unroll
    for (int j = 0; j < 7; ++j) {
      const int q = p + j - 3;
      if (q >= 0 && q < kSEQ)
        acc += x[((size_t)b * kSEQ + q) * kD + c] * w_dw[j * kD + c];
    }
    vals[h] = acc; s += acc; s2 += acc * acc;
  }
  __shared__ float rs[256], rq[256];
  rs[t] = s; rq[t] = s2;
  __syncthreads();
  for (int o = 128; o > 0; o >>= 1) {
    if (t < o) { rs[t] += rs[t + o]; rq[t] += rq[t + o]; }
    __syncthreads();
  }
  const float mu  = rs[0] * (1.0f / kD);
  const float var = rq[0] * (1.0f / kD) - mu * mu;
  const float inv = rsqrtf(var + 1e-6f);
#pragma unroll
  for (int h = 0; h < 2; ++h) {
    const int c = t + h * 256;
    yA[(size_t)m * kD + c] = f2bf((vals[h] - mu) * inv * g[c] + beta[c]);
  }
}

// ---------------- GRN: sum over sequence of h2^2 per (b, channel) ----------------
__global__ __launch_bounds__(256)
void gx2_kernel(const unsigned short* __restrict__ h2, float* __restrict__ Gx2) {
  const int b = blockIdx.x;
  const int c = blockIdx.y * 256 + threadIdx.x;
  const unsigned short* pbase = h2 + (size_t)b * kSEQ * kDI + c;
  float s = 0.f;
  for (int p = 0; p < kSEQ; ++p) {
    const float v = bf2f(pbase[(size_t)p * kDI]);
    s += v * v;
  }
  Gx2[b * kDI + c] = s;
}

// scale[b,c] = 1 + g[c] * (sqrt(Gx2[b,c]) / (mean_c sqrt(Gx2[b,:]) + 1e-6))
__global__ __launch_bounds__(256)
void grn_scale_kernel(const float* __restrict__ Gx2, const float* __restrict__ g,
                      float* __restrict__ scale) {
  const int b = blockIdx.x, t = threadIdx.x;
  float gx[4]; float s = 0.f;
#pragma unroll
  for (int h = 0; h < 4; ++h) { gx[h] = sqrtf(Gx2[b * kDI + t + h * 256]); s += gx[h]; }
  __shared__ float rs[256];
  rs[t] = s; __syncthreads();
  for (int o = 128; o > 0; o >>= 1) { if (t < o) rs[t] += rs[t + o]; __syncthreads(); }
  const float mean = rs[0] * (1.0f / kDI);
#pragma unroll
  for (int h = 0; h < 4; ++h) {
    const int c = t + h * 256;
    scale[b * kDI + c] = 1.0f + g[c] * (gx[h] / (mean + 1e-6f));
  }
}

// h3 = h2*scale + grn_b  (in place, bf16)
__global__ __launch_bounds__(256)
void grn_apply_kernel(unsigned short* __restrict__ h2, const float* __restrict__ scale,
                      const float* __restrict__ gb) {
  const size_t idx = (size_t)blockIdx.x * 256 + threadIdx.x;   // over kM*kDI
  const int c = (int)(idx % kDI);
  const int b = (int)(idx / ((size_t)kSEQ * kDI));
  const float v = bf2f(h2[idx]);
  h2[idx] = f2bf(v * scale[b * kDI + c] + gb[c]);
}

// ---------------- weight cast+transpose: W[K][N] f32 -> Wt[N][K] bf16 ----------------
__global__ __launch_bounds__(256)
void wcast_kernel(const float* __restrict__ W, unsigned short* __restrict__ Wt,
                  int K, int N) {
  const int idx = blockIdx.x * 256 + threadIdx.x;
  const int k = idx / N, n = idx % N;
  Wt[(size_t)n * K + k] = f2bf(W[idx]);
}

// ---------------- mask counts per batch row ----------------
__global__ __launch_bounds__(256)
void count_kernel(const unsigned char* __restrict__ mask, const int* __restrict__ seqp,
                  int* __restrict__ counts) {
  const int b = blockIdx.x, t = threadIdx.x;
  const int tvalid = min(kNT, seqp[0]);
  int a = 0, v = 0;
  for (int p = t; p < kSEQ; p += 256) {
    const int mv = (mask[(size_t)b * kSEQ + p] != 0) ? 1 : 0;
    a += mv;
    v += (mv && (p < tvalid)) ? 1 : 0;
  }
  __shared__ int ra[256], rv[256];
  ra[t] = a; rv[t] = v; __syncthreads();
  for (int o = 128; o > 0; o >>= 1) {
    if (t < o) { ra[t] += ra[t + o]; rv[t] += rv[t + o]; }
    __syncthreads();
  }
  if (t == 0) { counts[2 * b] = ra[0]; counts[2 * b + 1] = rv[0]; }
}

// ---------------- avg upsample (valid prefix contiguous, as in reference setup) ----------------
__global__ __launch_bounds__(256)
void upsample_kernel(const float* __restrict__ x, const int* __restrict__ counts,
                     float* __restrict__ out) {
  const size_t idx = (size_t)blockIdx.x * 256 + threadIdx.x;   // over kB*kSEQ*kD
  const int c = (int)(idx % kD);
  const size_t m = idx / kD;
  const int t = (int)(m % kSEQ);
  const int b = (int)(m / kSEQ);
  const int audio = counts[2 * b];
  const int valid = counts[2 * b + 1];
  float v = 0.0f;
  if (valid > 0 && t < audio) {
    const long br  = (long)audio / valid;
    const long rem = (long)audio - br * valid;
    const long k   = (long)valid - rem;
    long j = ((long)t < k * br) ? (long)t / (br > 0 ? br : 1)
                                : k + ((long)t - k * br) / (br + 1);
    if (j < 0) j = 0;
    if (j > (long)valid - 1) j = (long)valid - 1;
    v = x[((size_t)b * kSEQ + j) * kD + c];    // order[j] == j: valid prefix contiguous
  }
  out[idx] = v;
}

// ---------------- host-side orchestration ----------------
extern "C" void kernel_launch(void* const* d_in, const int* in_sizes, int n_in,
                              void* d_out, int out_size, void* d_ws, size_t ws_size,
                              hipStream_t stream) {
  const int*   text  = (const int*)d_in[0];
  const int*   seqp  = (const int*)d_in[1];
  const unsigned char* amask = (const unsigned char*)d_in[2];
  const float* etab  = (const float*)d_in[3];
  const float* w_dw  = (const float*)d_in[4];
  const float* b_dw  = (const float*)d_in[5];
  const float* ln_g  = (const float*)d_in[6];
  const float* ln_b  = (const float*)d_in[7];
  const float* W1    = (const float*)d_in[8];
  const float* b1    = (const float*)d_in[9];
  const float* grn_g = (const float*)d_in[10];
  const float* grn_b = (const float*)d_in[11];
  const float* W2    = (const float*)d_in[12];
  const float* b2    = (const float*)d_in[13];
  float* outp = (float*)d_out;

  // workspace carve-up (~163 MB)
  char* ws = (char*)d_ws;
  size_t off = 0;
  float*          x   = (float*)(ws + off);          off += (size_t)kM * kD  * sizeof(float);
  unsigned short* yA  = (unsigned short*)(ws + off); off += (size_t)kM * kD  * sizeof(unsigned short);
  unsigned short* h2  = (unsigned short*)(ws + off); off += (size_t)kM * kDI * sizeof(unsigned short);
  unsigned short* W1t = (unsigned short*)(ws + off); off += (size_t)kDI * kD * sizeof(unsigned short);
  unsigned short* W2t = (unsigned short*)(ws + off); off += (size_t)kD * kDI * sizeof(unsigned short);
  float*          Gx2 = (float*)(ws + off);          off += (size_t)kB * kDI * sizeof(float);
  float*          scl = (float*)(ws + off);          off += (size_t)kB * kDI * sizeof(float);
  int*            cnt = (int*)(ws + off);            off += (size_t)kB * 2   * sizeof(int);
  (void)ws_size; (void)in_sizes; (void)n_in; (void)out_size;

  count_kernel<<<kB, 256, 0, stream>>>(amask, seqp, cnt);
  embed_kernel<<<(kM * kD) / 256, 256, 0, stream>>>(text, etab, seqp, x);

  for (int l = 0; l < kL; ++l) {
    wcast_kernel<<<(kD * kDI) / 256, 256, 0, stream>>>(W1 + (size_t)l * kD * kDI, W1t, kD, kDI);
    wcast_kernel<<<(kDI * kD) / 256, 256, 0, stream>>>(W2 + (size_t)l * kDI * kD, W2t, kDI, kD);

    dwconv_ln_kernel<<<kM, 256, 0, stream>>>(x, w_dw + (size_t)l * 7 * kD,
                                             b_dw + (size_t)l * kD,
                                             ln_g + (size_t)l * kD,
                                             ln_b + (size_t)l * kD, yA);

    // h2 = gelu(yA @ W1 + b1)   [32768 x 512 x 1024], bf16 WMMA, f32 acc
    wmma_gemm_kernel<kD, kDI, true>
        <<<dim3(kM / 128, kDI / 128), 256, 0, stream>>>(yA, W1t, b1 + (size_t)l * kDI,
                                                        h2, nullptr, nullptr, seqp);

    gx2_kernel<<<dim3(kB, kDI / 256), 256, 0, stream>>>(h2, Gx2);
    grn_scale_kernel<<<kB, 256, 0, stream>>>(Gx2, grn_g + (size_t)l * kDI, scl);
    grn_apply_kernel<<<((size_t)kM * kDI) / 256, 256, 0, stream>>>(h2, scl,
                                                                   grn_b + (size_t)l * kDI);

    // x = h3 @ W2 + b2 + x, pad rows zeroed   [32768 x 1024 x 512], in-place residual
    wmma_gemm_kernel<kDI, kD, false>
        <<<dim3(kM / 128, kD / 128), 256, 0, stream>>>(h2, W2t, b2 + (size_t)l * kD,
                                                       nullptr, x, x, seqp);
  }

  upsample_kernel<<<(kM * kD) / 256, 256, 0, stream>>>(x, cnt, outp);
}
